// TransformerLayerShard_10479720202252
// MI455X (gfx1250) — matrix operand order repaired and checked by reading, not compiled, not run on hardware
//
#include <hip/hip_runtime.h>

typedef __attribute__((ext_vector_type(16))) __bf16 v16bf;
typedef __attribute__((ext_vector_type(8)))  float  v8f;
typedef unsigned short ushort_t;

#define D_MODEL 2048
#define FF      8192
#define NTOK    4096      /* B*T */
#define NHEAD   16
#define DHEAD   128
#define TSEQ    2048
#define NBATCH  2

struct U32x8 { uint4 lo, hi; };

static __device__ __forceinline__ v16bf make_frag(uint4 lo, uint4 hi) {
    U32x8 t; t.lo = lo; t.hi = hi;
    return __builtin_bit_cast(v16bf, t);
}
static __device__ __forceinline__ ushort_t f2bf(float f) {
    unsigned u = __builtin_bit_cast(unsigned, f);
    u += 0x7FFFu + ((u >> 16) & 1u);
    return (ushort_t)(u >> 16);
}
static __device__ __forceinline__ float bf2f(ushort_t h) {
    return __builtin_bit_cast(float, ((unsigned)h) << 16);
}
static __device__ __forceinline__ float gelu_tanh(float x) {
    return 0.5f * x * (1.0f + tanhf(0.7978845608028654f * (x + 0.044715f * x * x * x)));
}

// Issue one CDNA5 async global->LDS 128-bit copy (per-lane). Tracked by ASYNCcnt.
static __device__ __forceinline__ void async_b128(unsigned lds_off, const ushort_t* gptr) {
    asm volatile("global_load_async_to_lds_b128 %0, %1, off"
                 :: "v"(lds_off), "v"((unsigned long long)(size_t)gptr)
                 : "memory");
}
static __device__ __forceinline__ void wait_async0() {
    asm volatile("s_wait_asynccnt 0" ::: "memory");
}
static __device__ __forceinline__ unsigned lds_addr(const void* p) {
    return (unsigned)(size_t)p;   // LDS aperture keeps allocation-relative offset in addr[31:0]
}

// ---------------- elementwise helpers ----------------
__global__ void copy_f32(const float* __restrict__ src, float* __restrict__ dst, size_t n) {
    size_t i = (size_t)blockIdx.x * blockDim.x + threadIdx.x;
    size_t stride = (size_t)gridDim.x * blockDim.x;
    for (; i < n; i += stride) dst[i] = src[i];
}

// ---------------- fp32 [K][N] -> bf16 transposed [N][K] (LDS-tiled) ----------------
__global__ __launch_bounds__(256) void cvt_transpose_bf16(const float* __restrict__ src,
                                                          ushort_t* __restrict__ dst,
                                                          int K, int N) {
    __shared__ float tile[32][33];
    const int k0 = blockIdx.y * 32;
    const int n0 = blockIdx.x * 32;
    const int tx = threadIdx.x;       // 0..31
    const int ty = threadIdx.y;       // 0..7
    #pragma unroll
    for (int i = 0; i < 4; ++i) {
        const int kk = ty + i * 8;
        tile[kk][tx] = src[(size_t)(k0 + kk) * N + n0 + tx];
    }
    __syncthreads();
    #pragma unroll
    for (int i = 0; i < 4; ++i) {
        const int nn = ty + i * 8;
        dst[(size_t)(n0 + nn) * K + k0 + tx] = f2bf(tile[tx][nn]);
    }
}

// ---------------- LayerNorm: fp32 in -> bf16 out ----------------
__global__ __launch_bounds__(256) void ln_kernel(const float* __restrict__ x,
                                                 const float* __restrict__ g,
                                                 const float* __restrict__ o,
                                                 ushort_t* __restrict__ xnb) {
    __shared__ float red[256];
    const int row = blockIdx.x;
    const int tid = threadIdx.x;
    const float* xr = x + (size_t)row * D_MODEL;
    float s = 0.f, ss = 0.f;
    for (int i = tid; i < D_MODEL; i += 256) { float v = xr[i]; s += v; ss += v * v; }
    red[tid] = s; __syncthreads();
    for (int st = 128; st > 0; st >>= 1) { if (tid < st) red[tid] += red[tid + st]; __syncthreads(); }
    const float mean = red[0] * (1.0f / D_MODEL);
    __syncthreads();
    red[tid] = ss; __syncthreads();
    for (int st = 128; st > 0; st >>= 1) { if (tid < st) red[tid] += red[tid + st]; __syncthreads(); }
    const float var = red[0] * (1.0f / D_MODEL) - mean * mean;
    const float rstd = rsqrtf(var + 1e-5f);
    for (int i = tid; i < D_MODEL; i += 256) {
        float v = (xr[i] - mean) * rstd * g[i] + o[i];
        xnb[(size_t)row * D_MODEL + i] = f2bf(v);
    }
}

// ---------------- RoPE in-place on bf16 Q or K ----------------
__global__ void rope_kernel(ushort_t* __restrict__ qk) {
    const size_t total = (size_t)NTOK * NHEAD * 32;
    size_t idx = (size_t)blockIdx.x * blockDim.x + threadIdx.x;
    if (idx >= total) return;
    const int j   = (int)(idx & 31);
    const int h   = (int)((idx >> 5) & 15);
    const int tok = (int)(idx >> 9);
    const int t   = tok & (TSEQ - 1);
    const float freq = (float)t * __expf(-(float)j * 0.28782313662425572f);
    const float sn = __sinf(freq), cs = __cosf(freq);
    size_t a = (size_t)tok * D_MODEL + h * DHEAD + 2 * j;
    float x0 = bf2f(qk[a]), x1 = bf2f(qk[a + 1]);
    qk[a]     = f2bf(x0 * cs - x1 * sn);
    qk[a + 1] = f2bf(x1 * cs + x0 * sn);
}

// ---------------- bf16 WMMA GEMM, async-LDS double-buffered ----------------
// C[M,N] = act( A[M,K] @ W^T + bias ) ; A row-major [M][K], Bt row-major [N][K]
// mode 0: store bf16 to outH ; mode 1: outF += result
#define BM 128
#define BN 128
#define BK 32
__global__ __launch_bounds__(256) void gemm_bf16(const ushort_t* __restrict__ A,
                                                 const ushort_t* __restrict__ Bt,
                                                 int M, int N, int K,
                                                 const float* __restrict__ bias,
                                                 int act, int mode,
                                                 float* __restrict__ outF,
                                                 ushort_t* __restrict__ outH) {
    __shared__ __align__(16) ushort_t As[2][BM][BK];   // 16 KB
    __shared__ __align__(16) ushort_t Bs[2][BN][BK];   // 16 KB
    const int tid  = threadIdx.x;
    const int lane = tid & 31;
    const int wave = tid >> 5;          // 0..7
    const int wr   = wave >> 1;         // 0..3 : 32-row group
    const int wc   = wave & 1;          // 0..1 : 64-col group
    const int bm0  = blockIdx.y * BM;
    const int bn0  = blockIdx.x * BN;
    const int kh   = lane >> 4;
    const int ln16 = lane & 15;

    const int srow = tid >> 1;          // 0..127
    const int scol = (tid & 1) * 16;    // 0 or 16

    v8f acc[2][4] = {};

    auto stage = [&](int buf, int k0) {
        const ushort_t* ga = A  + (size_t)(bm0 + srow) * K + k0 + scol;
        const ushort_t* gb = Bt + (size_t)(bn0 + srow) * K + k0 + scol;
        unsigned la = lds_addr(&As[buf][srow][scol]);
        unsigned lb = lds_addr(&Bs[buf][srow][scol]);
        async_b128(la,       ga);
        async_b128(la + 16u, ga + 8);
        async_b128(lb,       gb);
        async_b128(lb + 16u, gb + 8);
    };

    stage(0, 0);
    wait_async0();
    __syncthreads();

    const int nsteps = K / BK;
    for (int s = 0; s < nsteps; ++s) {
        const int cur = s & 1;
        if (s + 1 < nsteps) stage(cur ^ 1, (s + 1) * BK);   // overlap with compute

        v16bf af[2], bfv[4];
        #pragma unroll
        for (int i = 0; i < 2; ++i) {
            const int row = wr * 32 + i * 16 + ln16;
            af[i] = make_frag(*(const uint4*)(&As[cur][row][kh * 8]),
                              *(const uint4*)(&As[cur][row][16 + kh * 8]));
        }
        #pragma unroll
        for (int j = 0; j < 4; ++j) {
            const int n = wc * 64 + j * 16 + ln16;
            bfv[j] = make_frag(*(const uint4*)(&Bs[cur][n][kh * 8]),
                               *(const uint4*)(&Bs[cur][n][16 + kh * 8]));
        }
        #pragma unroll
        for (int i = 0; i < 2; ++i)
            #pragma unroll
            for (int j = 0; j < 4; ++j)
                acc[i][j] = __builtin_amdgcn_wmma_f32_16x16x32_bf16(
                    false, af[i], false, bfv[j], (short)0, acc[i][j], false, false);

        wait_async0();      // next-tile async copies have landed
        __syncthreads();
    }

    // epilogue: reg r holds row m = r + 8*(lane/16), col = lane%16
    #pragma unroll
    for (int i = 0; i < 2; ++i) {
        #pragma unroll
        for (int j = 0; j < 4; ++j) {
            const int col = bn0 + wc * 64 + j * 16 + ln16;
            const float bvadd = bias ? bias[col] : 0.0f;
            #pragma unroll
            for (int r = 0; r < 8; ++r) {
                const int row = bm0 + wr * 32 + i * 16 + r + 8 * kh;
                float v = acc[i][j][r] + bvadd;
                if (act) v = gelu_tanh(v);
                const size_t idx = (size_t)row * N + col;
                if (mode == 0) outH[idx] = f2bf(v);
                else           outF[idx] += v;
            }
        }
    }
}

// ---------------- flash attention: one wave per (b, h, 16-query tile) ----------------
__global__ __launch_bounds__(32) void attn_kernel(const ushort_t* __restrict__ Qb,
                                                  const ushort_t* __restrict__ Kb,
                                                  const ushort_t* __restrict__ Vb,
                                                  const float* __restrict__ abias,
                                                  ushort_t* __restrict__ ctx) {
    __shared__ __align__(16) ushort_t Kblk[32][DHEAD];   // [key][dim]
    __shared__ __align__(16) ushort_t Vt[DHEAD][32];     // [dim][key]
    __shared__ __align__(16) ushort_t Pbuf[16][32];      // probabilities (bf16)
    const int lane  = threadIdx.x;
    const int qt0   = blockIdx.x * 16;
    const int h     = blockIdx.y;
    const int b     = blockIdx.z;
    const int hbase = h * DHEAD;
    const int kh    = lane >> 4;
    const int ln16  = lane & 15;
    const float sc  = 0.08838834764831845f;   // 1/sqrt(128)
    const float MB  = 20.7232658369464f;      // |log(1e-9)| mask bias of reference

    v16bf qf[4];
    const size_t qrow = (size_t)(b * TSEQ + qt0 + ln16) * D_MODEL + hbase;
    #pragma unroll
    for (int c = 0; c < 4; ++c) {
        uint4 lo = *(const uint4*)(Qb + qrow + c * 32 + kh * 8);
        uint4 hi = *(const uint4*)(Qb + qrow + c * 32 + 16 + kh * 8);
        qf[c] = make_frag(lo, hi);
    }

    float m_run[8], l_run[8];
    v8f acc[8] = {};
    #pragma unroll
    for (int r = 0; r < 8; ++r) { m_run[r] = -1e30f; l_run[r] = 0.0f; }

    const int qhi = qt0 + 15;
    const int nkb = (qhi >> 5) + 1;

    for (int kb = 0; kb < nkb; ++kb) {
        const int kstart = kb * 32;
        {
            const size_t krow = (size_t)(b * TSEQ + kstart + lane) * D_MODEL + hbase;
            #pragma unroll
            for (int i = 0; i < 16; ++i) {
                uint4 kv = *(const uint4*)(Kb + krow + i * 8);
                *(uint4*)(&Kblk[lane][i * 8]) = kv;
                uint4 vv = *(const uint4*)(Vb + krow + i * 8);
                const ushort_t* vp = (const ushort_t*)&vv;
                #pragma unroll
                for (int j = 0; j < 8; ++j) Vt[i * 8 + j][lane] = vp[j];
            }
        }
        __syncthreads();

        v8f s0 = {}, s1 = {};
        #pragma unroll
        for (int c = 0; c < 4; ++c) {
            uint4 k0lo = *(const uint4*)(&Kblk[ln16][c * 32 + kh * 8]);
            uint4 k0hi = *(const uint4*)(&Kblk[ln16][c * 32 + 16 + kh * 8]);
            uint4 k1lo = *(const uint4*)(&Kblk[16 + ln16][c * 32 + kh * 8]);
            uint4 k1hi = *(const uint4*)(&Kblk[16 + ln16][c * 32 + 16 + kh * 8]);
            s0 = __builtin_amdgcn_wmma_f32_16x16x32_bf16(false, qf[c], false, make_frag(k0lo, k0hi),
                                                         (short)0, s0, false, false);
            s1 = __builtin_amdgcn_wmma_f32_16x16x32_bf16(false, qf[c], false, make_frag(k1lo, k1hi),
                                                         (short)0, s1, false, false);
        }

        #pragma unroll
        for (int r = 0; r < 8; ++r) {
            const int q  = qt0 + r + 8 * kh;
            const int k0 = kstart + ln16;
            const int k1 = kstart + 16 + ln16;
            float e0 = s0[r] * sc + (k0 <= q ? MB : -MB) + abias[k0];
            float e1 = s1[r] * sc + (k1 <= q ? MB : -MB) + abias[k1];
            float mx = fmaxf(e0, e1);
            #pragma unroll
            for (int o = 1; o < 16; o <<= 1) mx = fmaxf(mx, __shfl_xor(mx, o, 32));
            const float nm = fmaxf(m_run[r], mx);
            const float scale = __expf(m_run[r] - nm);
            const float p0 = __expf(e0 - nm);
            const float p1 = __expf(e1 - nm);
            float rs = p0 + p1;
            #pragma unroll
            for (int o = 1; o < 16; o <<= 1) rs += __shfl_xor(rs, o, 32);
            l_run[r] = l_run[r] * scale + rs;
            m_run[r] = nm;
            #pragma unroll
            for (int c = 0; c < 8; ++c) acc[c][r] *= scale;
            Pbuf[r + 8 * kh][ln16]      = f2bf(p0);
            Pbuf[r + 8 * kh][16 + ln16] = f2bf(p1);
        }
        __syncthreads();

        uint4 plo = *(const uint4*)(&Pbuf[ln16][kh * 8]);
        uint4 phi = *(const uint4*)(&Pbuf[ln16][16 + kh * 8]);
        v16bf pf = make_frag(plo, phi);
        #pragma unroll
        for (int c = 0; c < 8; ++c) {
            uint4 vlo = *(const uint4*)(&Vt[c * 16 + ln16][kh * 8]);
            uint4 vhi = *(const uint4*)(&Vt[c * 16 + ln16][16 + kh * 8]);
            acc[c] = __builtin_amdgcn_wmma_f32_16x16x32_bf16(false, pf, false, make_frag(vlo, vhi),
                                                             (short)0, acc[c], false, false);
        }
        __syncthreads();
    }

    #pragma unroll
    for (int r = 0; r < 8; ++r) {
        const float inv = 1.0f / l_run[r];
        const size_t row = (size_t)(b * TSEQ + qt0 + r + 8 * kh) * D_MODEL + hbase;
        #pragma unroll
        for (int c = 0; c < 8; ++c)
            ctx[row + c * 16 + ln16] = f2bf(acc[c][r] * inv);
    }
}

// ---------------- host-side orchestration ----------------
extern "C" void kernel_launch(void* const* d_in, const int* in_sizes, int n_in,
                              void* d_out, int out_size, void* d_ws, size_t ws_size,
                              hipStream_t stream) {
    const float* x      = (const float*)d_in[0];
    const float* abias  = (const float*)d_in[1];
    const float* lng    = (const float*)d_in[2];
    const float* lno    = (const float*)d_in[3];
    const float* Wq     = (const float*)d_in[4];
    const float* Wk     = (const float*)d_in[5];
    const float* Wv     = (const float*)d_in[6];
    const float* Wo     = (const float*)d_in[7];
    const float* W1     = (const float*)d_in[8];
    const float* b1     = (const float*)d_in[9];
    const float* W2     = (const float*)d_in[10];
    const float* b2     = (const float*)d_in[11];
    float* out = (float*)d_out;

    const size_t SZ_W   = (size_t)D_MODEL * D_MODEL;
    const size_t SZ_W1  = (size_t)D_MODEL * FF;
    const size_t SZ_ACT = (size_t)NTOK * D_MODEL;
    const size_t SZ_H   = (size_t)NTOK * FF;

    ushort_t* p   = (ushort_t*)d_ws;
    ushort_t* Wqt = p; p += SZ_W;     // transposed bf16 weights: [N][K]
    ushort_t* Wkt = p; p += SZ_W;
    ushort_t* Wvt = p; p += SZ_W;
    ushort_t* Wot = p; p += SZ_W;
    ushort_t* W1t = p; p += SZ_W1;
    ushort_t* W2t = p; p += SZ_W1;
    ushort_t* xnb = p; p += SZ_ACT;
    ushort_t* Qb  = p; p += SZ_ACT;
    ushort_t* Kbf = p; p += SZ_ACT;
    ushort_t* Vbf = p; p += SZ_ACT;
    ushort_t* ctx = p; p += SZ_ACT;
    ushort_t* h1b = p; p += SZ_H;

    // 1. weights -> bf16, transposed to [N][K]
    dim3 tb(32, 8);
    cvt_transpose_bf16<<<dim3(D_MODEL / 32, D_MODEL / 32), tb, 0, stream>>>(Wq, Wqt, D_MODEL, D_MODEL);
    cvt_transpose_bf16<<<dim3(D_MODEL / 32, D_MODEL / 32), tb, 0, stream>>>(Wk, Wkt, D_MODEL, D_MODEL);
    cvt_transpose_bf16<<<dim3(D_MODEL / 32, D_MODEL / 32), tb, 0, stream>>>(Wv, Wvt, D_MODEL, D_MODEL);
    cvt_transpose_bf16<<<dim3(D_MODEL / 32, D_MODEL / 32), tb, 0, stream>>>(Wo, Wot, D_MODEL, D_MODEL);
    cvt_transpose_bf16<<<dim3(FF / 32, D_MODEL / 32), tb, 0, stream>>>(W1, W1t, D_MODEL, FF);
    cvt_transpose_bf16<<<dim3(D_MODEL / 32, FF / 32), tb, 0, stream>>>(W2, W2t, FF, D_MODEL);

    // 2. LayerNorm
    ln_kernel<<<NTOK, 256, 0, stream>>>(x, lng, lno, xnb);

    // 3. QKV projections
    dim3 gQKV(D_MODEL / BN, NTOK / BM);
    gemm_bf16<<<gQKV, 256, 0, stream>>>(xnb, Wqt, NTOK, D_MODEL, D_MODEL, nullptr, 0, 0, nullptr, Qb);
    gemm_bf16<<<gQKV, 256, 0, stream>>>(xnb, Wkt, NTOK, D_MODEL, D_MODEL, nullptr, 0, 0, nullptr, Kbf);
    gemm_bf16<<<gQKV, 256, 0, stream>>>(xnb, Wvt, NTOK, D_MODEL, D_MODEL, nullptr, 0, 0, nullptr, Vbf);

    // 4. RoPE on Q, K
    const size_t npairs = (size_t)NTOK * NHEAD * 32;
    rope_kernel<<<(unsigned)((npairs + 255) / 256), 256, 0, stream>>>(Qb);
    rope_kernel<<<(unsigned)((npairs + 255) / 256), 256, 0, stream>>>(Kbf);

    // 5. flash attention
    dim3 gAttn(TSEQ / 16, NHEAD, NBATCH);
    attn_kernel<<<gAttn, 32, 0, stream>>>(Qb, Kbf, Vbf, abias, ctx);

    // 6. residual init: out = x
    copy_f32<<<8192, 256, 0, stream>>>(x, out, SZ_ACT);

    // 7. out += ctx @ Wo
    gemm_bf16<<<gQKV, 256, 0, stream>>>(ctx, Wot, NTOK, D_MODEL, D_MODEL, nullptr, 0, 1, out, nullptr);

    // 8. h1 = gelu(xn @ W1 + b1)
    dim3 gFF1(FF / BN, NTOK / BM);
    gemm_bf16<<<gFF1, 256, 0, stream>>>(xnb, W1t, NTOK, FF, D_MODEL, b1, 1, 0, nullptr, h1b);

    // 9. out += h1 @ W2 + b2
    gemm_bf16<<<gQKV, 256, 0, stream>>>(h1b, W2t, NTOK, D_MODEL, FF, b2, 0, 1, out, nullptr);
}
